// DIN_Attention_1975684956807
// MI455X (gfx1250) — matrix operand order repaired and checked by reading, not compiled, not run on hardware
//
#include <hip/hip_runtime.h>

typedef __attribute__((ext_vector_type(16))) __bf16 v16bf;
typedef __attribute__((ext_vector_type(8)))  float  v8f;

#define B_   2048
#define S_   200
#define SP_  208   // S padded to 13*16
#define D_   128
#define H_   64
#define KTOT 256   // folded K: [h | t*h]
#define PSTR 66    // part[] row stride (floats): bank = (2s+n)&63, conflict-free

static __device__ __forceinline__ unsigned short bf16bits(float f) {
    union { __bf16 b; unsigned short u; } cv;
    cv.b = (__bf16)f;
    return cv.u;
}
static __device__ __forceinline__ unsigned packbf(float a, float b) {
    return (unsigned)bf16bits(a) | ((unsigned)bf16bits(b) << 16);
}

// ---- Prep 1: wt[n][k] bf16, n<64, k<256. k<128: (W1_h - W1_d); k>=128: W1_p
__global__ __launch_bounds__(256) void din_prep_weights(const float* __restrict__ W1,
                                                        __bf16* __restrict__ wt) {
    for (int idx = blockIdx.x * blockDim.x + threadIdx.x; idx < H_ * KTOT;
         idx += gridDim.x * blockDim.x) {
        int n = idx >> 8;
        int k = idx & 255;
        float v;
        if (k < 128) v = W1[(128 + k) * H_ + n] - W1[(256 + k) * H_ + n];
        else         v = W1[(384 + (k - 128)) * H_ + n];
        wt[n * KTOT + k] = (__bf16)v;
    }
}

// ---- Prep 2: tA[b][n] = b1[n] + sum_k t[b][k]*(W1_t + W1_d)[k][n]
__global__ __launch_bounds__(64) void din_prep_ta(const float* __restrict__ tgt,
                                                  const float* __restrict__ W1,
                                                  const float* __restrict__ b1,
                                                  float* __restrict__ tA) {
    const int b = blockIdx.x, n = threadIdx.x;
    const float* t = tgt + (size_t)b * D_;
    float acc = b1[n];
    for (int k = 0; k < D_; ++k)
        acc = fmaf(t[k], W1[k * H_ + n] + W1[(256 + k) * H_ + n], acc);
    tA[b * H_ + n] = acc;
}

// ---- Main: one workgroup per batch row. 128 threads = 4 waves (wave32).
__global__ __launch_bounds__(128) void din_attn_main(const float* __restrict__ tgt,
                                                     const float* __restrict__ hist,
                                                     const int* __restrict__ mask,
                                                     const float* __restrict__ W2,
                                                     const float* __restrict__ b2,
                                                     const __bf16* __restrict__ wt,
                                                     const float* __restrict__ tA,
                                                     float* __restrict__ out) {
    extern __shared__ char smem[];
    // amat[s][k] bf16: k<128 -> h[s][k]; k>=128 -> t[k-128]*h[s][k-128]
    __bf16* amat = (__bf16*)smem;                       // [SP_][KTOT]   106,496 B
    float*  part = (float*)(amat + SP_ * KTOT);         // [SP_][PSTR]    54,912 B
    float*  wbuf = part + SP_ * PSTR;                   // [SP_]             832 B

    const int b    = blockIdx.x;
    const int tid  = threadIdx.x;
    const int wave = tid >> 5;
    const int lane = tid & 31;

    // ---------------- Phase 1: build folded A matrix in LDS (fp32 -> bf16)
    {
        const float4* h4 = (const float4*)(hist + (size_t)b * S_ * D_);
        const float4* t4 = (const float4*)(tgt + (size_t)b * D_);
        for (int i = tid; i < S_ * D_ / 4; i += 128) {
            float4 v = h4[i];
            int row = i >> 5, c4 = i & 31;      // c4*4 = feature index
            float4 t = t4[c4];                  // L1/L2-resident broadcast
            char* rb = (char*)amat + row * 512;
            *(uint2*)(rb + c4 * 8) =
                make_uint2(packbf(v.x, v.y), packbf(v.z, v.w));
            *(uint2*)(rb + 256 + c4 * 8) =
                make_uint2(packbf(v.x * t.x, v.y * t.y), packbf(v.z * t.z, v.w * t.w));
        }
        for (int i = tid; i < (SP_ - S_) * KTOT / 4; i += 128) {   // zero pad rows
            int row = S_ + (i >> 6), c4 = i & 63;
            *(uint2*)((char*)amat + row * 512 + c4 * 8) = make_uint2(0u, 0u);
        }
    }
    __syncthreads();

    // ---------------- Phase 2: part[s][n] = relu((A @ Wt)[s][n] + tA[n]) * W2[n]
    {
        const int n0 = wave * 16;
        const int n  = n0 + (lane & 15);

        // Preload this wave's B fragments (16 cols, K=256 in 8 steps of 32).
        // 16-bit B layout: lanes 0-15 -> K kb..kb+7 & kb+16..kb+23, lanes 16-31 -> +8
        // (kb = ks*32); matches the A layout.
        union BU { v16bf v; uint4 q[2]; } bfr[8];
        {
            const int kofB = (lane < 16) ? 0 : 8;
            const char* wrow = (const char*)wt + n * 512 + kofB * 2;
            #pragma unroll
            for (int ks = 0; ks < 8; ++ks) {
                bfr[ks].q[0] = *(const uint4*)(wrow + ks * 64);
                bfr[ks].q[1] = *(const uint4*)(wrow + ks * 64 + 32);
            }
        }
        const float tAv = tA[b * H_ + n];
        const float w2v = W2[n];
        const int rofs = (lane >= 16) ? 8 : 0;

        for (int m = 0; m < SP_ / 16; ++m) {
            const int row = m * 16 + (lane & 15);
            // 16-bit A layout: lane<16 -> K kb0..kb0+7 & kb0+16..kb0+23 (kb0=ks*32),
            // lane>=16 -> +8. Each half is 16 contiguous bytes -> ds_load_b128.
            const char* rb = (const char*)amat + row * 512 + ((lane < 16) ? 0 : 16);
            v8f c = {};
            #pragma unroll
            for (int ks = 0; ks < 8; ++ks) {
                union AU { v16bf v; uint4 q[2]; } a;
                a.q[0] = *(const uint4*)(rb + ks * 64);
                a.q[1] = *(const uint4*)(rb + ks * 64 + 32);
                c = __builtin_amdgcn_wmma_f32_16x16x32_bf16(false, a.v, false, bfr[ks].v,
                                                            (short)0, c, false, false);
            }
            // C layout: VGPR j: lanes 0-15 -> M=j, lanes 16-31 -> M=j+8; N=lane%16.
            // Store scaled partials; summed over n in phase 3 (no shuffles/atomics).
            #pragma unroll
            for (int j = 0; j < 8; ++j) {
                float p = fmaxf(c[j] + tAv, 0.0f) * w2v;
                part[(m * 16 + j + rofs) * PSTR + n] = p;
            }
        }
    }
    __syncthreads();

    // ---------------- Phase 3: logits -> mask -> softmax over S
    {
        const float b2v = b2[0];
        const int s0i = tid, s1i = tid + 128;
        float w0 = -1e9f, w1 = -1e9f;
        if (s0i < S_) {
            float acc = 0.0f;
            const float2* pr = (const float2*)(part + s0i * PSTR);
            #pragma unroll
            for (int i = 0; i < H_ / 2; ++i) { float2 v = pr[i]; acc += v.x + v.y; }
            float mm = (float)mask[(size_t)b * S_ + s0i];
            w0 = (acc + b2v) * mm - 1e9f * (1.0f - mm);
        }
        if (s1i < S_) {
            float acc = 0.0f;
            const float2* pr = (const float2*)(part + s1i * PSTR);
            #pragma unroll
            for (int i = 0; i < H_ / 2; ++i) { float2 v = pr[i]; acc += v.x + v.y; }
            float mm = (float)mask[(size_t)b * S_ + s1i];
            w1 = (acc + b2v) * mm - 1e9f * (1.0f - mm);
        }
        wbuf[s0i] = w0;
        if (s1i < SP_) wbuf[s1i] = w1;
        __syncthreads();
        float mx = -1e30f;
        for (int s = 0; s < SP_; ++s) mx = fmaxf(mx, wbuf[s]);
        float e0 = __expf(w0 - mx), e1 = __expf(w1 - mx);
        __syncthreads();
        wbuf[s0i] = e0;
        if (s1i < SP_) wbuf[s1i] = e1;
        __syncthreads();
        float sum = 0.0f;
        for (int s = 0; s < SP_; ++s) sum += wbuf[s];
        float inv = 1.0f / sum;
        __syncthreads();
        wbuf[s0i] = e0 * inv;
        if (s1i < SP_) wbuf[s1i] = e1 * inv;
        __syncthreads();
    }

    // ---------------- Phase 4: out[b][d] = sum_s w[s] * h[b][s][d] (LDS bf16)
    {
        const int d = tid;     // blockDim==128==D_
        float acc = 0.0f;
        for (int s = 0; s < S_; ++s)
            acc = fmaf(wbuf[s], (float)amat[s * KTOT + d], acc);
        out[(size_t)b * D_ + d] = acc;
    }
}

extern "C" void kernel_launch(void* const* d_in, const int* in_sizes, int n_in,
                              void* d_out, int out_size, void* d_ws, size_t ws_size,
                              hipStream_t stream) {
    (void)in_sizes; (void)n_in; (void)out_size; (void)ws_size;
    const float* tgt  = (const float*)d_in[0];
    const float* hist = (const float*)d_in[1];
    const int*   mask = (const int*)d_in[2];
    const float* W1   = (const float*)d_in[3];
    const float* b1   = (const float*)d_in[4];
    const float* W2   = (const float*)d_in[5];
    const float* b2   = (const float*)d_in[6];
    float* out = (float*)d_out;

    // Workspace carve: wt bf16 [64][256] (32 KB) + tA f32 [2048][64] (512 KB)
    __bf16* wt = (__bf16*)d_ws;
    float*  tA = (float*)((char*)d_ws + 64 * KTOT * sizeof(__bf16));

    din_prep_weights<<<64, 256, 0, stream>>>(W1, wt);
    din_prep_ta<<<B_, H_, 0, stream>>>(tgt, W1, b1, tA);

    // LDS: 106,496 (amat) + 54,912 (part) + 832 (wbuf) = 162,240 B
    const size_t smem = (size_t)SP_ * KTOT * 2 + (size_t)SP_ * PSTR * 4 + SP_ * 4;
    din_attn_main<<<B_, 128, smem, stream>>>(tgt, hist, mask, W2, b2, wt, tA, out);
}
